// DeformAttnFusionNet_79328045957391
// MI455X (gfx1250) — matrix operand order repaired and checked by reading, not compiled, not run on hardware
//
#include <hip/hip_runtime.h>
#include <hip/hip_bf16.h>

// ---------------------------------------------------------------------------
// DeformAttn fused pipeline for gfx1250 (MI455X, wave32, WMMA)
//   K1: v      = value @ W_value + b_value           -> ws (layout [B,NH,HW,HD])
//   K2: oa     = query @ [W_offset|W_attn] + bias    -> ws ([B*Q, 96])
//   K3: gather = softmax + bilinear sample + point-agg -> ws ([B*Q, 128])
//   K4: out    = agg @ W_out + b_out                 -> d_out
// GEMMs use v_wmma_f32_16x16x32_bf16 with hi/lo bf16 split-precision
// (3 WMMAs per K32 chunk) for ~fp32 accuracy at tensor-pipe rates.
// ---------------------------------------------------------------------------

#define BATCH 2
#define NQ    40000
#define CDIM  128
#define NHEAD 8
#define NPTS  4
#define FH    90
#define FW    160
#define HWSZ  (FH * FW)      // 14400
#define HD    16

typedef __bf16 v16bf __attribute__((ext_vector_type(16)));
typedef float  v8f   __attribute__((ext_vector_type(8)));

union FragBF {
    v16bf        v;
    unsigned int u[8];
};
union FragC {
    v8f   v;
    float f[8];
};

__device__ __forceinline__ unsigned short f2bf_rne(float x) {
    unsigned int u = __float_as_uint(x);
    unsigned int r = u + 0x7FFFu + ((u >> 16) & 1u);
    return (unsigned short)(r >> 16);
}
__device__ __forceinline__ float bf2f(unsigned short b) {
    return __uint_as_float(((unsigned int)b) << 16);
}

// ---------------------------------------------------------------------------
// Generic WMMA GEMM: Out[M, N] = A[M,128] @ W[128,N] + bias
// N = NTILES*16.  W is the concat of W0 (N0 cols) and W1 (N1 cols).
// VALUE_MODE: scatter output to [B, NH, HW, HD] layout (N must be 128).
// Block = 256 threads (8 waves); each block computes 128 rows.
// ---------------------------------------------------------------------------
template <int NTILES, bool VALUE_MODE>
__global__ __launch_bounds__(256)
void gemm_wmma_k(const float* __restrict__ A,
                 const float* __restrict__ W0, int N0,
                 const float* __restrict__ W1, int N1,
                 const float* __restrict__ b0,
                 const float* __restrict__ b1,
                 float* __restrict__ Out) {
    constexpr int N = NTILES * 16;
    constexpr int FRAG_STRIDE = 20;                 // uints per lane-fragment (padded)
    __shared__ unsigned int lds_w[NTILES * 4 * 32 * FRAG_STRIDE];

    const int tid  = threadIdx.x;
    const int lane = tid & 31;
    const int wave = tid >> 5;
    const bool laneHi = (lane >= 16);

    // ---- Pre-swizzle weights into LDS in WMMA B-fragment layout (hi+lo bf16)
    // B 32x16 bf16 layout: lane l (<16): col n=l,    e0..7 -> k 0..7,  e8..15 -> k 16..23
    //                      lane l (>=16): col n=l-16, e0..7 -> k 8..15, e8..15 -> k 24..31
    const int total = NTILES * 4 * 32;
    for (int t = tid; t < total; t += 256) {
        const int fl   = t & 31;
        const int kc   = (t >> 5) & 3;
        const int nt   = t >> 7;
        const int n    = nt * 16 + (fl & 15);
        const int kbas = kc * 32 + ((fl >= 16) ? 8 : 0);
        unsigned int* dst = &lds_w[t * FRAG_STRIDE];
#pragma unroll
        for (int j = 0; j < 8; ++j) {
            const int k0 = kbas + 2 * j + ((j >= 4) ? 8 : 0);
            const int k1 = k0 + 1;
            const float w0 = (n < N0) ? W0[(size_t)k0 * N0 + n] : W1[(size_t)k1 * 0 + (size_t)k0 * N1 + (n - N0)];
            const float w1 = (n < N0) ? W0[(size_t)k1 * N0 + n] : W1[(size_t)k1 * N1 + (n - N0)];
            const unsigned short h0 = f2bf_rne(w0);
            const unsigned short h1 = f2bf_rne(w1);
            const unsigned short l0 = f2bf_rne(w0 - bf2f(h0));
            const unsigned short l1 = f2bf_rne(w1 - bf2f(h1));
            dst[j]     = (unsigned int)h0 | ((unsigned int)h1 << 16);
            dst[8 + j] = (unsigned int)l0 | ((unsigned int)l1 << 16);
        }
    }
    __syncthreads();

    const int rowBase = blockIdx.x * 128 + wave * 16;
    const int aRow    = rowBase + (lane & 15);

    FragC acc[NTILES];
#pragma unroll
    for (int nt = 0; nt < NTILES; ++nt)
#pragma unroll
        for (int r = 0; r < 8; ++r) acc[nt].f[r] = 0.0f;

#pragma unroll
    for (int kc = 0; kc < 4; ++kc) {
        // ---- Build A fragments (hi/lo) from global.
        // A 16x32 bf16 layout: lane l row m=l%16;
        //   l<16 : e0..7 -> k 0..7,  e8..15 -> k 16..23
        //   l>=16: e0..7 -> k 8..15, e8..15 -> k 24..31
        const float* ar = A + (size_t)aRow * 128 + kc * 32 + (laneHi ? 8 : 0);
        const float4 p0 = *reinterpret_cast<const float4*>(ar);
        const float4 p1 = *reinterpret_cast<const float4*>(ar + 4);
        const float4 p2 = *reinterpret_cast<const float4*>(ar + 16);
        const float4 p3 = *reinterpret_cast<const float4*>(ar + 20);
        float e[16] = {p0.x, p0.y, p0.z, p0.w, p1.x, p1.y, p1.z, p1.w,
                       p2.x, p2.y, p2.z, p2.w, p3.x, p3.y, p3.z, p3.w};
        FragBF ah, al;
#pragma unroll
        for (int j = 0; j < 8; ++j) {
            const unsigned short h0 = f2bf_rne(e[2 * j]);
            const unsigned short h1 = f2bf_rne(e[2 * j + 1]);
            const unsigned short l0 = f2bf_rne(e[2 * j] - bf2f(h0));
            const unsigned short l1 = f2bf_rne(e[2 * j + 1] - bf2f(h1));
            ah.u[j] = (unsigned int)h0 | ((unsigned int)h1 << 16);
            al.u[j] = (unsigned int)l0 | ((unsigned int)l1 << 16);
        }

#pragma unroll
        for (int nt = 0; nt < NTILES; ++nt) {
            const unsigned int* fp = &lds_w[((nt * 4 + kc) * 32 + lane) * FRAG_STRIDE];
            const uint4* f4 = reinterpret_cast<const uint4*>(fp);
            FragBF bh, bl;
            uint4 t0 = f4[0], t1 = f4[1], t2 = f4[2], t3 = f4[3];
            bh.u[0] = t0.x; bh.u[1] = t0.y; bh.u[2] = t0.z; bh.u[3] = t0.w;
            bh.u[4] = t1.x; bh.u[5] = t1.y; bh.u[6] = t1.z; bh.u[7] = t1.w;
            bl.u[0] = t2.x; bl.u[1] = t2.y; bl.u[2] = t2.z; bl.u[3] = t2.w;
            bl.u[4] = t3.x; bl.u[5] = t3.y; bl.u[6] = t3.z; bl.u[7] = t3.w;

            acc[nt].v = __builtin_amdgcn_wmma_f32_16x16x32_bf16(
                false, ah.v, false, bh.v, (short)0, acc[nt].v, false, false);
            acc[nt].v = __builtin_amdgcn_wmma_f32_16x16x32_bf16(
                false, ah.v, false, bl.v, (short)0, acc[nt].v, false, false);
            acc[nt].v = __builtin_amdgcn_wmma_f32_16x16x32_bf16(
                false, al.v, false, bh.v, (short)0, acc[nt].v, false, false);
        }
    }

    // ---- Epilogue: bias + store.  C/D layout: lane l, reg r -> row r+(l>=16?8:0), col l%16
#pragma unroll
    for (int nt = 0; nt < NTILES; ++nt) {
        const int col = nt * 16 + (lane & 15);
        const float bv = (col < N0) ? b0[col] : b1[col - N0];
#pragma unroll
        for (int r = 0; r < 8; ++r) {
            const int gRow = rowBase + r + (laneHi ? 8 : 0);
            const float val = acc[nt].f[r] + bv;
            if (VALUE_MODE) {
                // gRow = b*HW + hw ; head = nt ; d = col % 16
                const int b  = gRow / HWSZ;
                const int hw = gRow - b * HWSZ;
                Out[(((size_t)(b * NHEAD + nt) * HWSZ) + hw) * HD + (lane & 15)] = val;
            } else {
                Out[(size_t)gRow * N + col] = val;
            }
        }
    }
}

// ---------------------------------------------------------------------------
// Gather kernel: one wave per query; lane = head*4 + point.
// softmax over the 4 points of each head (quad shfl_xor), bilinear sample
// from vproj [B,NH,HW,HD] (contiguous 64B rows), quad-reduce over points.
// ---------------------------------------------------------------------------
__global__ __launch_bounds__(256)
void deform_gather_k(const float* __restrict__ vproj,
                     const float* __restrict__ offsattn,
                     const float* __restrict__ ref,
                     float* __restrict__ agg) {
    const int wave = threadIdx.x >> 5;
    const int lane = threadIdx.x & 31;
    const int q = blockIdx.x * 8 + wave;          // 0 .. B*Q-1 (exact multiple)
    const int b = q / NQ;
    const int h = lane >> 2;
    const int p = lane & 3;

    const float* oa = offsattn + (size_t)q * 96;
    const float off_x = oa[h * 8 + p * 2 + 0];
    const float off_y = oa[h * 8 + p * 2 + 1];
    const float logit = oa[64 + h * 4 + p];

    const float rx = ref[(size_t)q * 2 + 0];
    const float ry = ref[(size_t)q * 2 + 1];

    // sampling position in pixel space (align_corners=False)
    const float x = rx * (float)FW + off_x - 0.5f;
    const float y = ry * (float)FH + off_y - 0.5f;

    // softmax over the 4 points in this head (lanes h*4 .. h*4+3)
    float m = logit;
    m = fmaxf(m, __shfl_xor(m, 1));
    m = fmaxf(m, __shfl_xor(m, 2));
    float ev = __expf(logit - m);
    float s = ev;
    s += __shfl_xor(s, 1);
    s += __shfl_xor(s, 2);
    const float aw = ev / s;

    const float xf = floorf(x), yf = floorf(y);
    const float wx = x - xf, wy = y - yf;
    const int x0 = (int)xf, y0 = (int)yf;

    const float* vhead = vproj + (size_t)(b * NHEAD + h) * HWSZ * HD;

    float4 a0 = {0.f, 0.f, 0.f, 0.f}, a1 = a0, a2 = a0, a3 = a0;

    auto corner = [&](int xi, int yi, float cw) {
        const bool valid = (xi >= 0) & (xi < FW) & (yi >= 0) & (yi < FH);
        const float w = aw * cw * (valid ? 1.0f : 0.0f);
        const int xc = min(max(xi, 0), FW - 1);
        const int yc = min(max(yi, 0), FH - 1);
        const float4* vp = reinterpret_cast<const float4*>(vhead + (size_t)(yc * FW + xc) * HD);
        const float4 v0 = vp[0], v1 = vp[1], v2 = vp[2], v3 = vp[3];
        a0.x += w * v0.x; a0.y += w * v0.y; a0.z += w * v0.z; a0.w += w * v0.w;
        a1.x += w * v1.x; a1.y += w * v1.y; a1.z += w * v1.z; a1.w += w * v1.w;
        a2.x += w * v2.x; a2.y += w * v2.y; a2.z += w * v2.z; a2.w += w * v2.w;
        a3.x += w * v3.x; a3.y += w * v3.y; a3.z += w * v3.z; a3.w += w * v3.w;
    };
    corner(x0,     y0,     (1.f - wx) * (1.f - wy));
    corner(x0 + 1, y0,     wx * (1.f - wy));
    corner(x0,     y0 + 1, (1.f - wx) * wy);
    corner(x0 + 1, y0 + 1, wx * wy);

    // reduce over the 4 points of the head (quad lanes)
    auto qr = [](float v) {
        v += __shfl_xor(v, 1);
        v += __shfl_xor(v, 2);
        return v;
    };
    a0.x = qr(a0.x); a0.y = qr(a0.y); a0.z = qr(a0.z); a0.w = qr(a0.w);
    a1.x = qr(a1.x); a1.y = qr(a1.y); a1.z = qr(a1.z); a1.w = qr(a1.w);
    a2.x = qr(a2.x); a2.y = qr(a2.y); a2.z = qr(a2.z); a2.w = qr(a2.w);
    a3.x = qr(a3.x); a3.y = qr(a3.y); a3.z = qr(a3.z); a3.w = qr(a3.w);

    if (p == 0) {
        float4* dst = reinterpret_cast<float4*>(agg + (size_t)q * CDIM + h * HD);
        dst[0] = a0; dst[1] = a1; dst[2] = a2; dst[3] = a3;
    }
}

// ---------------------------------------------------------------------------
extern "C" void kernel_launch(void* const* d_in, const int* in_sizes, int n_in,
                              void* d_out, int out_size, void* d_ws, size_t ws_size,
                              hipStream_t stream) {
    (void)in_sizes; (void)n_in; (void)out_size; (void)ws_size;
    const float* query    = (const float*)d_in[0];   // [B,Q,128]
    const float* value    = (const float*)d_in[1];   // [B,HW,128]
    const float* refpts   = (const float*)d_in[2];   // [B,Q,2]
    const float* W_value  = (const float*)d_in[3];   // [128,128]
    const float* b_value  = (const float*)d_in[4];
    const float* W_offset = (const float*)d_in[5];   // [128,64]
    const float* b_offset = (const float*)d_in[6];
    const float* W_attn   = (const float*)d_in[7];   // [128,32]
    const float* b_attn   = (const float*)d_in[8];
    const float* W_out    = (const float*)d_in[9];   // [128,128]
    const float* b_out    = (const float*)d_in[10];
    float* out = (float*)d_out;                      // [B,Q,128]

    // workspace partition (floats)
    float* vproj    = (float*)d_ws;                    // B*NH*HW*HD = 3,686,400
    float* offsattn = vproj + (size_t)BATCH * NHEAD * HWSZ * HD;
    float* agg      = offsattn + (size_t)BATCH * NQ * 96;  // B*Q*96 = 7,680,000

    // K1: value projection -> [B,NH,HW,HD].  M = 28800 rows -> 225 blocks.
    gemm_wmma_k<8, true><<<(BATCH * HWSZ) / 128, 256, 0, stream>>>(
        value, W_value, 128, nullptr, 0, b_value, nullptr, vproj);

    // K2: offsets+attn logits -> [B*Q, 96].  M = 80000 -> 625 blocks.
    gemm_wmma_k<6, false><<<(BATCH * NQ) / 128, 256, 0, stream>>>(
        query, W_offset, 64, W_attn, 32, b_offset, b_attn, offsattn);

    // K3: softmax + bilinear gather + point aggregation -> [B*Q, 128]
    deform_gather_k<<<(BATCH * NQ) / 8, 256, 0, stream>>>(
        vproj, offsattn, refpts, agg);

    // K4: output projection -> d_out.  M = 80000 -> 625 blocks.
    gemm_wmma_k<8, false><<<(BATCH * NQ) / 128, 256, 0, stream>>>(
        agg, W_out, 128, nullptr, 0, b_out, nullptr, out);
}
